// GroupedExperts_46334107189526
// MI455X (gfx1250) — compile-verified, matrix operand order checked by reading, not run on hardware
//
#include <hip/hip_runtime.h>

// ---------- types ----------
typedef __attribute__((ext_vector_type(16))) __bf16 v16bf;
typedef __attribute__((ext_vector_type(2)))  __bf16 v2bf;
typedef __attribute__((ext_vector_type(8)))  float  v8f;
typedef __attribute__((ext_vector_type(4)))  float  f32x4;
typedef __attribute__((ext_vector_type(4)))  unsigned int u32x4;

static constexpr int E_ = 8;
static constexpr int T_ = 2048;
static constexpr int D_ = 2048;
static constexpr int H_ = 4096;

// fp32 -> bf16 round-to-nearest-even (scalar fallback)
__device__ __forceinline__ unsigned short f32_to_bf16(float f) {
  unsigned int u = __float_as_uint(f);
  u += 0x7FFFu + ((u >> 16) & 1u);
  return (unsigned short)(u >> 16);
}

// Pack two fp32 into a bf16x2 dword (low = a, high = b).
__device__ __forceinline__ unsigned int pack_bf16(float a, float b) {
#if __has_builtin(__builtin_amdgcn_cvt_pk_bf16_f32)
  union { v2bf v; unsigned int u; } c;
  c.v = __builtin_amdgcn_cvt_pk_bf16_f32(a, b);
  return c.u;
#else
  return (unsigned int)f32_to_bf16(a) | ((unsigned int)f32_to_bf16(b) << 16);
#endif
}

// Async 16B Global -> LDS copy (CDNA5 GLOBAL_LOAD_ASYNC_TO_LDS_B128, GV mode).
// vdst VGPR carries the LDS byte address; generic pointers to __shared__ data
// hold that byte address in their low 32 bits (ISA 10.2 aperture layout).
__device__ __forceinline__ void async_copy_b128(void* lds_dst, const void* gsrc) {
  unsigned lds_off = (unsigned)(size_t)lds_dst;
  asm volatile("global_load_async_to_lds_b128 %0, %1, off"
               :: "v"(lds_off), "v"(gsrc) : "memory");
}

__device__ __forceinline__ void wait_asynccnt0() {
#if __has_builtin(__builtin_amdgcn_s_wait_asynccnt)
  __builtin_amdgcn_s_wait_asynccnt(0);
#else
  asm volatile("s_wait_asynccnt 0x0" ::: "memory");
#endif
}

// Load a 16x32 (MxK) bf16 A-fragment (or 32x16 KxN B-fragment stored
// K-contiguous per row) from an LDS tile laid out [row][32 k], row stride 32.
// Per CDNA5 ISA 16-bit A layout: lane = half*16 + m; VGPR j holds K pair:
//   j<4 : K = half*8 + 2j          j>=4 : K = 16 + half*8 + 2(j-4)
__device__ __forceinline__ v16bf load_frag_b16(const unsigned short* lds,
                                               int rowbase, int lane) {
  const int r = rowbase + (lane & 15);
  const int half4 = ((lane >> 4) & 1) * 4;   // dword offset within 16-dword row
  union { unsigned int u[8]; v16bf v; } f;
  const unsigned int* p = (const unsigned int*)(lds + r * 32);
#pragma unroll
  for (int j = 0; j < 4; ++j) {
    f.u[j]     = p[half4 + j];
    f.u[4 + j] = p[8 + half4 + j];
  }
  return f.v;
}

// =====================================================================
// Kernel 1: h[e,t,n] = silu(x@w1) * (x@w3), bf16 output into workspace
// Grid: (H/128, T/128, E), 256 threads (8 waves: 4 M-waves x 2 N-waves)
// =====================================================================
__global__ __launch_bounds__(256) void swiglu_up_kernel(
    const float* __restrict__ x, const float* __restrict__ w1,
    const float* __restrict__ w3, unsigned short* __restrict__ hbuf) {
  __shared__ unsigned short lds_a [128 * 32];
  __shared__ unsigned short lds_b1[128 * 32];
  __shared__ unsigned short lds_b3[128 * 32];

  const int tid  = threadIdx.x;
  const int lane = tid & 31;
  const int wid  = tid >> 5;
  const int wm   = wid & 3;   // wave row:   32 rows each
  const int wn   = wid >> 2;  // wave col:   64 cols each

  const int e  = blockIdx.z;
  const int tm = blockIdx.y;
  const int tn = blockIdx.x;

  const float* xe  = x  + (size_t)e * T_ * D_ + (size_t)tm * 128 * D_;
  const float* w1e = w1 + (size_t)e * D_ * H_ + (size_t)tn * 128;
  const float* w3e = w3 + (size_t)e * D_ * H_ + (size_t)tn * 128;

  v8f acc1[2][4], acc3[2][4];
#pragma unroll
  for (int i = 0; i < 2; ++i)
#pragma unroll
    for (int j = 0; j < 4; ++j) { acc1[i][j] = {}; acc3[i][j] = {}; }

  for (int kk = 0; kk < D_ / 32; ++kk) {
    const int k0 = kk * 32;
    __syncthreads();
    // A tile: 128 rows x 32 K of x, fp32 -> bf16 packed pairs, K-contiguous
#pragma unroll
    for (int i = 0; i < 4; ++i) {
      int q   = tid + i * 256;      // 0..1023 float4's
      int row = q >> 3;
      int c4  = (q & 7) * 4;
      f32x4 v = *(const f32x4*)(xe + (size_t)row * D_ + k0 + c4);
      unsigned int* dst = (unsigned int*)&lds_a[row * 32 + c4];
      dst[0] = pack_bf16(v[0], v[1]);
      dst[1] = pack_bf16(v[2], v[3]);
    }
    // B tiles: 32 K rows x 128 N of w1/w3 -> [n][k], packed k-pairs
#pragma unroll
    for (int i = 0; i < 2; ++i) {
      int q  = tid + i * 256;       // 0..511
      int rp = (q >> 5) * 2;        // k row-pair base
      int c4 = (q & 31) * 4;        // n group of 4
      size_t g0 = (size_t)(k0 + rp) * H_ + c4;
      f32x4 a1 = *(const f32x4*)(w1e + g0);
      f32x4 b1 = *(const f32x4*)(w1e + g0 + H_);
      f32x4 a3 = *(const f32x4*)(w3e + g0);
      f32x4 b3 = *(const f32x4*)(w3e + g0 + H_);
#pragma unroll
      for (int j = 0; j < 4; ++j) {
        *(unsigned int*)&lds_b1[(c4 + j) * 32 + rp] = pack_bf16(a1[j], b1[j]);
        *(unsigned int*)&lds_b3[(c4 + j) * 32 + rp] = pack_bf16(a3[j], b3[j]);
      }
    }
    // Prefetch next K-tile into near caches while this stage computes
    if (kk + 1 < D_ / 32) {
      const int kn = k0 + 32;
#pragma unroll
      for (int i = 0; i < 4; ++i) {
        int q = tid + i * 256;
        __builtin_prefetch(xe + (size_t)(q >> 3) * D_ + kn + (q & 7) * 4, 0, 3);
      }
#pragma unroll
      for (int i = 0; i < 2; ++i) {
        int q = tid + i * 256;
        size_t g0 = (size_t)(kn + (q >> 5) * 2) * H_ + (q & 31) * 4;
        __builtin_prefetch(w1e + g0, 0, 3);
        __builtin_prefetch(w3e + g0, 0, 3);
      }
    }
    __syncthreads();

    v16bf a0 = load_frag_b16(lds_a, wm * 32,      lane);
    v16bf a1 = load_frag_b16(lds_a, wm * 32 + 16, lane);
#pragma unroll
    for (int nf = 0; nf < 4; ++nf) {
      v16bf b1f = load_frag_b16(lds_b1, wn * 64 + nf * 16, lane);
      v16bf b3f = load_frag_b16(lds_b3, wn * 64 + nf * 16, lane);
      acc1[0][nf] = __builtin_amdgcn_wmma_f32_16x16x32_bf16(
          false, a0, false, b1f, (short)0, acc1[0][nf], false, false);
      acc1[1][nf] = __builtin_amdgcn_wmma_f32_16x16x32_bf16(
          false, a1, false, b1f, (short)0, acc1[1][nf], false, false);
      acc3[0][nf] = __builtin_amdgcn_wmma_f32_16x16x32_bf16(
          false, a0, false, b3f, (short)0, acc3[0][nf], false, false);
      acc3[1][nf] = __builtin_amdgcn_wmma_f32_16x16x32_bf16(
          false, a1, false, b3f, (short)0, acc3[1][nf], false, false);
    }
  }

  // Epilogue: h = silu(g) * u, bf16 -> workspace.
  // C layout: lane = half*16 + n; VGPR r holds row = half*8 + r.
  unsigned short* hout = hbuf + (size_t)e * T_ * H_;
  const int coln = lane & 15;
  const int rh   = (lane >> 4) * 8;
#pragma unroll
  for (int mf = 0; mf < 2; ++mf)
#pragma unroll
    for (int nf = 0; nf < 4; ++nf)
#pragma unroll
      for (int r = 0; r < 8; ++r) {
        int row = tm * 128 + wm * 32 + mf * 16 + rh + r;
        int col = tn * 128 + wn * 64 + nf * 16 + coln;
        float g  = acc1[mf][nf][r];
        float u  = acc3[mf][nf][r];
        // silu(g) = g * rcp(1 + exp(-g)): v_exp + v_rcp, no IEEE divide
        float s  = __builtin_amdgcn_rcpf(1.f + __expf(-g));
        hout[(size_t)row * H_ + col] = f32_to_bf16(g * s * u);
      }
}

// =====================================================================
// Kernel 2: out[e,t,d] = h @ w2, fp32 accumulate/store
// Grid: (D/128, T/128, E), 256 threads
// =====================================================================
__global__ __launch_bounds__(256) void down_proj_kernel(
    const unsigned short* __restrict__ hbuf, const float* __restrict__ w2,
    float* __restrict__ out) {
  __shared__ unsigned short lds_a[128 * 32];
  __shared__ unsigned short lds_b[128 * 32];

  const int tid  = threadIdx.x;
  const int lane = tid & 31;
  const int wid  = tid >> 5;
  const int wm   = wid & 3;
  const int wn   = wid >> 2;

  const int e  = blockIdx.z;
  const int tm = blockIdx.y;
  const int tn = blockIdx.x;

  const unsigned short* he = hbuf + (size_t)e * T_ * H_ + (size_t)tm * 128 * H_;
  const float* w2e = w2 + (size_t)e * H_ * D_ + (size_t)tn * 128;

  v8f acc[2][4];
#pragma unroll
  for (int i = 0; i < 2; ++i)
#pragma unroll
    for (int j = 0; j < 4; ++j) acc[i][j] = {};

  for (int kk = 0; kk < H_ / 32; ++kk) {
    const int k0 = kk * 32;
    __syncthreads();
    // A tile: 128 x 32 bf16 already in workspace: pure copy -> use the
    // async Global->LDS engine (no VGPR staging, tracked by ASYNCcnt).
#pragma unroll
    for (int i = 0; i < 2; ++i) {
      int q   = tid + i * 256;
      int row = q >> 2;
      int seg = (q & 3) * 8;        // 8 bf16 = 16 bytes
      async_copy_b128(&lds_a[row * 32 + seg],
                      he + (size_t)row * H_ + k0 + seg);
    }
    // B tile: w2 32 K rows x 128 N -> [n][k], packed k-pairs
#pragma unroll
    for (int i = 0; i < 2; ++i) {
      int q  = tid + i * 256;
      int rp = (q >> 5) * 2;
      int c4 = (q & 31) * 4;
      size_t g0 = (size_t)(k0 + rp) * D_ + c4;
      f32x4 a = *(const f32x4*)(w2e + g0);
      f32x4 b = *(const f32x4*)(w2e + g0 + D_);
#pragma unroll
      for (int j = 0; j < 4; ++j)
        *(unsigned int*)&lds_b[(c4 + j) * 32 + rp] = pack_bf16(a[j], b[j]);
    }
    // Prefetch next K-tile
    if (kk + 1 < H_ / 32) {
      const int kn = k0 + 32;
#pragma unroll
      for (int i = 0; i < 2; ++i) {
        int q = tid + i * 256;
        __builtin_prefetch(he + (size_t)(q >> 2) * H_ + kn + (q & 3) * 8, 0, 3);
        size_t g0 = (size_t)(kn + (q >> 5) * 2) * D_ + (q & 31) * 4;
        __builtin_prefetch(w2e + g0, 0, 3);
      }
    }
    wait_asynccnt0();   // our async A-tile transfers landed in LDS
    __syncthreads();    // publish to all waves

    v16bf a0 = load_frag_b16(lds_a, wm * 32,      lane);
    v16bf a1 = load_frag_b16(lds_a, wm * 32 + 16, lane);
#pragma unroll
    for (int nf = 0; nf < 4; ++nf) {
      v16bf bf = load_frag_b16(lds_b, wn * 64 + nf * 16, lane);
      acc[0][nf] = __builtin_amdgcn_wmma_f32_16x16x32_bf16(
          false, a0, false, bf, (short)0, acc[0][nf], false, false);
      acc[1][nf] = __builtin_amdgcn_wmma_f32_16x16x32_bf16(
          false, a1, false, bf, (short)0, acc[1][nf], false, false);
    }
  }

  float* oe = out + (size_t)e * T_ * D_;
  const int coln = lane & 15;
  const int rh   = (lane >> 4) * 8;
#pragma unroll
  for (int mf = 0; mf < 2; ++mf)
#pragma unroll
    for (int nf = 0; nf < 4; ++nf)
#pragma unroll
      for (int r = 0; r < 8; ++r) {
        int row = tm * 128 + wm * 32 + mf * 16 + rh + r;
        int col = tn * 128 + wn * 64 + nf * 16 + coln;
        oe[(size_t)row * D_ + col] = acc[mf][nf][r];
      }
}

extern "C" void kernel_launch(void* const* d_in, const int* in_sizes, int n_in,
                              void* d_out, int out_size, void* d_ws, size_t ws_size,
                              hipStream_t stream) {
  (void)in_sizes; (void)n_in; (void)out_size; (void)ws_size;
  const float* x  = (const float*)d_in[0];
  const float* w1 = (const float*)d_in[1];
  const float* w2 = (const float*)d_in[2];
  const float* w3 = (const float*)d_in[3];
  float* out = (float*)d_out;
  unsigned short* hbuf = (unsigned short*)d_ws;   // E*T*H bf16 = 128 MB

  dim3 blk(256);
  dim3 g1(H_ / 128, T_ / 128, E_);   // 32 x 16 x 8
  swiglu_up_kernel<<<g1, blk, 0, stream>>>(x, w1, w3, hbuf);
  dim3 g2(D_ / 128, T_ / 128, E_);   // 16 x 16 x 8
  down_proj_kernel<<<g2, blk, 0, stream>>>(hbuf, w2, out);
}
